// GQNN_55602646614393
// MI455X (gfx1250) — matrix operand order repaired and checked by reading, not compiled, not run on hardware
//
#include <hip/hip_runtime.h>
#include <math.h>

typedef __attribute__((ext_vector_type(16))) _Float16 v16h;
typedef __attribute__((ext_vector_type(8)))  _Float16 v8h;
typedef __attribute__((ext_vector_type(8)))  float    v8f;

#define N_NODES 100000
#define FDIM 128

// ---------------------------------------------------------------- utilities

__global__ void gs_zero_kernel(float* __restrict__ p, int n) {
    int i = blockIdx.x * blockDim.x + threadIdx.x;
    if (i < n) p[i] = 0.0f;
}

// degree count: one thread per edge
__global__ void gs_count_kernel(const int* __restrict__ dst, float* __restrict__ cnt, int E) {
    int e = blockIdx.x * blockDim.x + threadIdx.x;
    if (e < E) unsafeAtomicAdd(&cnt[dst[e]], 1.0f);
}

// in-place: cnt -> 1/max(cnt,1)
__global__ void gs_inv_kernel(float* __restrict__ cnt, int n) {
    int i = blockIdx.x * blockDim.x + threadIdx.x;
    if (i < n) cnt[i] = 1.0f / fmaxf(cnt[i], 1.0f);
}

// scatter-add: 32 threads per edge, one float4 per thread
__global__ void gs_scatter_kernel(const float* __restrict__ X,
                                  const int* __restrict__ src,
                                  const int* __restrict__ dst,
                                  float* __restrict__ agg, int E) {
    long tid = (long)blockIdx.x * blockDim.x + threadIdx.x;
    int e = (int)(tid >> 5);
    if (e >= E) return;
    int q = (int)(tid & 31);
    int s = src[e], d = dst[e];
    float4 v = ((const float4*)(X + (long)s * FDIM))[q];
    float* a = agg + (long)d * FDIM + q * 4;
    unsafeAtomicAdd(a + 0, v.x);
    unsafeAtomicAdd(a + 1, v.y);
    unsafeAtomicAdd(a + 2, v.z);
    unsafeAtomicAdd(a + 3, v.w);
}

// W[k][n] (f32, 128x128) -> WT[n][k] (f16)
__global__ void gs_wt_kernel(const float* __restrict__ W, _Float16* __restrict__ WT) {
    int i = blockIdx.x * blockDim.x + threadIdx.x;   // 16384 threads
    int k = i >> 7, n = i & 127;
    WT[n * FDIM + k] = (_Float16)W[k * FDIM + n];
}

// ---------------------------------------------------------------- SAGE GEMM
// H[16 rows x 128 cols per block] = relu( (agg*inv) @ Wl + X @ Wr + b )
// 256 threads = 8 waves; wave w computes the 16x16 tile at column w*16.
__global__ __launch_bounds__(256)
void gs_gemm_kernel(const float* __restrict__ agg, const float* __restrict__ X,
                    const float* __restrict__ inv,
                    const _Float16* __restrict__ WlT, const _Float16* __restrict__ WrT,
                    const float* __restrict__ bias, float* __restrict__ H) {
    __shared__ __align__(16) _Float16 AsA[16][136];   // scaled agg tile (f16)
    __shared__ __align__(16) _Float16 AsX[16][136];   // x tile (f16)

    const int t = threadIdx.x;
    const int row0 = blockIdx.x * 16;

    // ---- stage A tiles: thread t loads row t>>4, cols (t&15)*8 .. +7
    {
        const int r = t >> 4, c0 = (t & 15) * 8;
        const long g = (long)(row0 + r) * FDIM + c0;
        const float s = inv[row0 + r];
        float4 a0 = *(const float4*)(agg + g);
        float4 a1 = *(const float4*)(agg + g + 4);
        float4 x0 = *(const float4*)(X + g);
        float4 x1 = *(const float4*)(X + g + 4);
        AsA[r][c0 + 0] = (_Float16)(a0.x * s); AsA[r][c0 + 1] = (_Float16)(a0.y * s);
        AsA[r][c0 + 2] = (_Float16)(a0.z * s); AsA[r][c0 + 3] = (_Float16)(a0.w * s);
        AsA[r][c0 + 4] = (_Float16)(a1.x * s); AsA[r][c0 + 5] = (_Float16)(a1.y * s);
        AsA[r][c0 + 6] = (_Float16)(a1.z * s); AsA[r][c0 + 7] = (_Float16)(a1.w * s);
        AsX[r][c0 + 0] = (_Float16)x0.x; AsX[r][c0 + 1] = (_Float16)x0.y;
        AsX[r][c0 + 2] = (_Float16)x0.z; AsX[r][c0 + 3] = (_Float16)x0.w;
        AsX[r][c0 + 4] = (_Float16)x1.x; AsX[r][c0 + 5] = (_Float16)x1.y;
        AsX[r][c0 + 6] = (_Float16)x1.z; AsX[r][c0 + 7] = (_Float16)x1.w;
    }
    __syncthreads();

    const int w    = t >> 5;            // wave id -> column tile
    const int lane = t & 31;
    const int M     = lane & 15;        // A row within tile
    const int kbase = (lane >> 4) * 16; // A K sub-run
    const int kb    = (lane >> 4) * 16; // B K sub-run
    const int ncol  = w * 16 + (lane & 15);

    // seed accumulator with bias (same N for all 8 M rows of this lane)
    const float bv = bias[ncol];
    v8f c;
    #pragma unroll
    for (int r = 0; r < 8; ++r) c[r] = bv;

    #pragma unroll
    for (int ks = 0; ks < 4; ++ks) {
        const int kO = ks * 32;
        v8h alo = *(const v8h*)&AsA[M][kO + kbase];
        v8h ahi = *(const v8h*)&AsA[M][kO + kbase + 8];
        v16h aA = __builtin_shufflevector(alo, ahi, 0,1,2,3,4,5,6,7,8,9,10,11,12,13,14,15);
        v8h xlo = *(const v8h*)&AsX[M][kO + kbase];
        v8h xhi = *(const v8h*)&AsX[M][kO + kbase + 8];
        v16h aX = __builtin_shufflevector(xlo, xhi, 0,1,2,3,4,5,6,7,8,9,10,11,12,13,14,15);
        v16h bl = *(const v16h*)(WlT + (long)ncol * FDIM + kO + kb);
        v16h br = *(const v16h*)(WrT + (long)ncol * FDIM + kO + kb);
        c = __builtin_amdgcn_wmma_f32_16x16x32_f16(false, aA, false, bl, (short)0, c, false, false);
        c = __builtin_amdgcn_wmma_f32_16x16x32_f16(false, aX, false, br, (short)0, c, false, false);
    }

    // relu + store: c[r] -> H[row0 + (lane>>4)*8 + r][ncol]
    const int mbase = row0 + (lane >> 4) * 8;
    #pragma unroll
    for (int r = 0; r < 8; ++r)
        H[(long)(mbase + r) * FDIM + ncol] = fmaxf(c[r], 0.0f);
}

// ---------------------------------------------------------------- heads
// wave per node: preds = h@Wp + bp ; diffs = sigmoid(h@Wd + bd)
// out[0..N) = preds - diffs ; out[N..2N) = preds + diffs
__global__ __launch_bounds__(256)
void gs_head_kernel(const float* __restrict__ H,
                    const float* __restrict__ Wp, const float* __restrict__ bp,
                    const float* __restrict__ Wd, const float* __restrict__ bd,
                    float* __restrict__ out) {
    const int node = blockIdx.x * 8 + (threadIdx.x >> 5);
    const int lane = threadIdx.x & 31;
    float4 h = ((const float4*)(H + (long)node * FDIM))[lane];
    float4 p = ((const float4*)Wp)[lane];
    float4 d = ((const float4*)Wd)[lane];
    float sp = h.x * p.x + h.y * p.y + h.z * p.z + h.w * p.w;
    float sd = h.x * d.x + h.y * d.y + h.z * d.z + h.w * d.w;
    #pragma unroll
    for (int off = 16; off > 0; off >>= 1) {
        sp += __shfl_xor(sp, off, 32);
        sd += __shfl_xor(sd, off, 32);
    }
    if (lane == 0) {
        float preds = sp + bp[0];
        float z     = sd + bd[0];
        float diffs = 1.0f / (1.0f + expf(-z));
        out[node]           = preds - diffs;
        out[N_NODES + node] = preds + diffs;
    }
}

// ---------------------------------------------------------------- launch

extern "C" void kernel_launch(void* const* d_in, const int* in_sizes, int n_in,
                              void* d_out, int out_size, void* d_ws, size_t ws_size,
                              hipStream_t stream) {
    const float* x   = (const float*)d_in[0];
    const int*   ei  = (const int*)d_in[1];
    const float* Wl1 = (const float*)d_in[2];
    const float* Wr1 = (const float*)d_in[3];
    const float* b1  = (const float*)d_in[4];
    const float* Wl2 = (const float*)d_in[5];
    const float* Wr2 = (const float*)d_in[6];
    const float* b2  = (const float*)d_in[7];
    const float* Wp  = (const float*)d_in[8];
    const float* bp  = (const float*)d_in[9];
    const float* Wd  = (const float*)d_in[10];
    const float* bd  = (const float*)d_in[11];
    float* out = (float*)d_out;

    const int E = in_sizes[1] / 2;
    const int* src = ei;
    const int* dst = ei + E;

    // workspace layout
    char* ws = (char*)d_ws;
    const size_t NB = (size_t)N_NODES * FDIM * sizeof(float);  // 51.2 MB
    float* agg = (float*)(ws);
    float* h1  = (float*)(ws + NB);
    float* h2  = (float*)(ws + 2 * NB);
    float* inv = (float*)(ws + 3 * NB);
    _Float16* wlt1 = (_Float16*)(ws + 3 * NB + (1u << 20));
    _Float16* wrt1 = wlt1 + FDIM * FDIM;
    _Float16* wlt2 = wrt1 + FDIM * FDIM;
    _Float16* wrt2 = wlt2 + FDIM * FDIM;

    const int NF = N_NODES * FDIM;                 // 12.8M
    const int zb_feat = (NF + 255) / 256;
    const int zb_node = (N_NODES + 255) / 256;
    const long sthreads = (long)E * 32;
    const int sb = (int)((sthreads + 255) / 256);

    // degrees -> inv
    gs_zero_kernel<<<zb_node, 256, 0, stream>>>(inv, N_NODES);
    gs_count_kernel<<<(E + 255) / 256, 256, 0, stream>>>(dst, inv, E);
    gs_inv_kernel<<<zb_node, 256, 0, stream>>>(inv, N_NODES);

    // weight transpose+convert (f32 -> f16, transposed for contiguous B frags)
    gs_wt_kernel<<<64, 256, 0, stream>>>(Wl1, wlt1);
    gs_wt_kernel<<<64, 256, 0, stream>>>(Wr1, wrt1);
    gs_wt_kernel<<<64, 256, 0, stream>>>(Wl2, wlt2);
    gs_wt_kernel<<<64, 256, 0, stream>>>(Wr2, wrt2);

    // layer 1
    gs_zero_kernel<<<zb_feat, 256, 0, stream>>>(agg, NF);
    gs_scatter_kernel<<<sb, 256, 0, stream>>>(x, src, dst, agg, E);
    gs_gemm_kernel<<<N_NODES / 16, 256, 0, stream>>>(agg, x, inv, wlt1, wrt1, b1, h1);

    // layer 2
    gs_zero_kernel<<<zb_feat, 256, 0, stream>>>(agg, NF);
    gs_scatter_kernel<<<sb, 256, 0, stream>>>(h1, src, dst, agg, E);
    gs_gemm_kernel<<<N_NODES / 16, 256, 0, stream>>>(agg, h1, inv, wlt2, wrt2, b2, h2);

    // heads
    gs_head_kernel<<<N_NODES / 8, 256, 0, stream>>>(h2, Wp, bp, Wd, bd, out);
}
